// Attention_55070070669572
// MI455X (gfx1250) — compile-verified, matrix operand order checked by reading, not run on hardware
//
#include <hip/hip_runtime.h>

// ---------------------------------------------------------------------------
// Problem constants (match reference)
// ---------------------------------------------------------------------------
#define B_    2
#define S_    2048
#define D_    4096
#define H_    32
#define KVH_  8
#define HD_   128            // head dim
#define NREP_ 4              // H_/KVH_
#define KVD_  (KVH_*HD_)     // 1024

typedef __attribute__((ext_vector_type(16))) __bf16 v16bf;
typedef __attribute__((ext_vector_type(8)))  __bf16 v8bf;
typedef __attribute__((ext_vector_type(8)))  float  v8f;
typedef __attribute__((ext_vector_type(4)))  float  v4f;
typedef __attribute__((ext_vector_type(4)))  int    v4i;

// CDNA5 async global->LDS path (ASYNCcnt), guarded so compile stays safe.
#if __has_builtin(__builtin_amdgcn_global_load_async_to_lds_b128) && \
    __has_builtin(__builtin_amdgcn_s_wait_asynccnt)
#define USE_ASYNC 1
#else
#define USE_ASYNC 0
#endif
// Builtin signature (from hipcc diagnostic): param0 = int4 addrspace(1)*,
// param1 = int4 addrspace(3)*, then imm offset, imm cpol.
#define GAS(p) ((__attribute__((address_space(1))) v4i*)(const void*)(p))
#define LAS(p) ((__attribute__((address_space(3))) v4i*)(void*)(p))

static __device__ __forceinline__ v8f wmma_bf16(v16bf a, v16bf b, v8f c) {
  return __builtin_amdgcn_wmma_f32_16x16x32_bf16(false, a, false, b,
                                                 (short)0, c, false, false);
}

// ---------------------------------------------------------------------------
// Within-16-lane xor shuffles: prefer VALU v_permlane16 over LDS bpermute.
// ---------------------------------------------------------------------------
#if __has_builtin(__builtin_amdgcn_permlane16)
static __device__ __forceinline__ float pl16(float x, unsigned lo, unsigned hi) {
  unsigned u = __builtin_bit_cast(unsigned, x);
  u = __builtin_amdgcn_permlane16(u, u, lo, hi, false, false);
  return __builtin_bit_cast(float, u);
}
static __device__ __forceinline__ float rowmax16(float v) {
  v = fmaxf(v, pl16(v, 0x67452301u, 0xEFCDAB89u));  // xor 1
  v = fmaxf(v, pl16(v, 0x54761032u, 0xDCFE98BAu));  // xor 2
  v = fmaxf(v, pl16(v, 0x32107654u, 0xBA98FEDCu));  // xor 4
  v = fmaxf(v, pl16(v, 0xFEDCBA98u, 0x76543210u));  // xor 8
  return v;
}
static __device__ __forceinline__ float rowsum16(float v) {
  v += pl16(v, 0x67452301u, 0xEFCDAB89u);
  v += pl16(v, 0x54761032u, 0xDCFE98BAu);
  v += pl16(v, 0x32107654u, 0xBA98FEDCu);
  v += pl16(v, 0xFEDCBA98u, 0x76543210u);
  return v;
}
#else
static __device__ __forceinline__ float rowmax16(float v) {
  #pragma unroll
  for (int m = 1; m < 16; m <<= 1) v = fmaxf(v, __shfl_xor(v, m, 32));
  return v;
}
static __device__ __forceinline__ float rowsum16(float v) {
  #pragma unroll
  for (int m = 1; m < 16; m <<= 1) v += __shfl_xor(v, m, 32);
  return v;
}
#endif

// ---------------------------------------------------------------------------
// fp32 -> bf16 bulk conversion (one-time; halves weight/activation traffic)
// ---------------------------------------------------------------------------
__global__ __launch_bounds__(256)
void cvt_bf16_kernel(const float* __restrict__ src, __bf16* __restrict__ dst,
                     int n8) {
  int i = blockIdx.x * blockDim.x + threadIdx.x;
  if (i >= n8) return;
  const v4f* s = (const v4f*)(src + (size_t)i * 8);
  v4f f0 = s[0], f1 = s[1];
  v8bf r;
  r[0] = (__bf16)f0[0]; r[1] = (__bf16)f0[1];
  r[2] = (__bf16)f0[2]; r[3] = (__bf16)f0[3];
  r[4] = (__bf16)f1[0]; r[5] = (__bf16)f1[1];
  r[6] = (__bf16)f1[2]; r[7] = (__bf16)f1[3];
  *(v8bf*)(dst + (size_t)i * 8) = r;
}

// ---------------------------------------------------------------------------
// GEMM: C[M,N] = A[M,K] @ W[N,K]^T, A/W bf16, out bf16 or f32.
// 128x128 block tile, BK=32, 8 waves; double-buffered async LDS staging.
// ---------------------------------------------------------------------------
template <bool OUT_BF16>
__global__ __launch_bounds__(256)
void gemm_bf16_kernel(const __bf16* __restrict__ A, const __bf16* __restrict__ W,
                      void* __restrict__ Cout, int M, int N, int K) {
  constexpr int BKP = 48;                 // row stride (halves): 96B = 3*32B
  __shared__ __bf16 As[2][128][BKP];
  __shared__ __bf16 Bs[2][128][BKP];

  const int tid   = threadIdx.x;
  const int lane  = tid & 31;
  const int w     = tid >> 5;
  const int ln    = lane & 15;
  const int lg    = lane >> 4;
  const int waveM = (w >> 2) * 64;        // 0 or 64
  const int waveN = (w & 3) * 32;         // 0,32,64,96
  const int blockN = blockIdx.x * 128;
  const int blockM = blockIdx.y * 128;

  v8f acc[4][2];
  #pragma unroll
  for (int i = 0; i < 4; ++i)
    #pragma unroll
    for (int j = 0; j < 2; ++j)
      #pragma unroll
      for (int e = 0; e < 8; ++e) acc[i][j][e] = 0.0f;

  // Stage one 128x32 bf16 tile of A and W: 512 16B chunks -> 2 per thread each.
  auto stage = [&](int k0, int buf) {
    #pragma unroll
    for (int i = 0; i < 2; ++i) {
      int cid = tid + i * 256;
      int row = cid >> 2;
      int cc  = (cid & 3) << 3;
      const __bf16* ga = A + (size_t)(blockM + row) * K + k0 + cc;
      const __bf16* gb = W + (size_t)(blockN + row) * K + k0 + cc;
#if USE_ASYNC
      __builtin_amdgcn_global_load_async_to_lds_b128(GAS(ga), LAS(&As[buf][row][cc]), 0, 0);
      __builtin_amdgcn_global_load_async_to_lds_b128(GAS(gb), LAS(&Bs[buf][row][cc]), 0, 0);
#else
      *(v8bf*)&As[buf][row][cc] = *(const v8bf*)ga;
      *(v8bf*)&Bs[buf][row][cc] = *(const v8bf*)gb;
#endif
    }
  };

  const int nk = K / 32;
  stage(0, 0);
  for (int kt = 0; kt < nk; ++kt) {
    const int buf = kt & 1;
    if (kt + 1 < nk) {
      stage((kt + 1) * 32, buf ^ 1);      // prefetch next tile into other buffer
#if USE_ASYNC
      __builtin_amdgcn_s_wait_asynccnt(4); // 4 outstanding = next tile only
#endif
    } else {
#if USE_ASYNC
      __builtin_amdgcn_s_wait_asynccnt(0);
#endif
    }
    __syncthreads();

    v16bf bf0 = *(const v16bf*)&Bs[buf][waveN +  0 + ln][lg * 16];
    v16bf bf1 = *(const v16bf*)&Bs[buf][waveN + 16 + ln][lg * 16];
    #pragma unroll
    for (int mt = 0; mt < 4; ++mt) {
      v16bf af = *(const v16bf*)&As[buf][waveM + mt * 16 + ln][lg * 16];
      acc[mt][0] = wmma_bf16(af, bf0, acc[mt][0]);
      acc[mt][1] = wmma_bf16(af, bf1, acc[mt][1]);
    }
    __syncthreads();                      // protect buf before it is re-staged
  }

  #pragma unroll
  for (int mt = 0; mt < 4; ++mt) {
    #pragma unroll
    for (int nt = 0; nt < 2; ++nt) {
      #pragma unroll
      for (int r = 0; r < 8; ++r) {
        int grow = blockM + waveM + mt * 16 + r + 8 * lg;
        int gcol = blockN + waveN + nt * 16 + ln;
        float val = acc[mt][nt][r];
        if (OUT_BF16)
          ((__bf16*)Cout)[(size_t)grow * N + gcol] = (__bf16)val;
        else
          ((float*)Cout)[(size_t)grow * N + gcol] = val;
      }
    }
  }
}

// ---------------------------------------------------------------------------
// RoPE on a bf16 buffer [rows, cols], interleaved pairs per head of 128.
// ---------------------------------------------------------------------------
__global__ void rope_kernel(__bf16* __restrict__ buf,
                            const float* __restrict__ cosb,
                            const float* __restrict__ sinb,
                            int rows, int cols) {
  int idx  = blockIdx.x * blockDim.x + threadIdx.x;
  int half = cols >> 1;
  if (idx >= rows * half) return;
  int m   = idx / half;
  int col = (idx - m * half) * 2;
  int p   = (col & (HD_ - 1)) >> 1;
  int sdx = m & (S_ - 1);
  float c = cosb[sdx * (HD_ / 2) + p];
  float s = sinb[sdx * (HD_ / 2) + p];
  size_t off = (size_t)m * cols + col;
  float a = (float)buf[off], b = (float)buf[off + 1];
  buf[off]     = (__bf16)(a * c - b * s);
  buf[off + 1] = (__bf16)(a * s + b * c);
}

// ---------------------------------------------------------------------------
// Flash attention: grid (S/128, H, B), 256 threads = 8 waves.
// Wave w owns 16 query rows; loops over 32-key blocks with online softmax.
// ---------------------------------------------------------------------------
__global__ __launch_bounds__(256)
void attn_kernel(const __bf16* __restrict__ q, const __bf16* __restrict__ k,
                 const __bf16* __restrict__ v, __bf16* __restrict__ o) {
  __shared__ __bf16 Ks[32][144];          // [key][hd]   stride 288B = 9*32B
  __shared__ __bf16 Vs[128][48];          // [hd][key]   stride  96B = 3*32B
  __shared__ __bf16 Ps[8][16][32];        // per-wave P staging (C->A relayout)

  const int tid   = threadIdx.x;
  const int lane  = tid & 31;
  const int w     = tid >> 5;
  const int ln    = lane & 15;
  const int lg    = lane >> 4;
  const int qtile = blockIdx.x;
  const int h     = blockIdx.y;
  const int b     = blockIdx.z;
  const int kvh   = h / NREP_;
  const int qrow0 = qtile * 128 + w * 16;
  const float scale = 0.08838834764831845f;   // 1/sqrt(128)

  // Q fragments held in registers: 16 rows x 128 cols bf16 (A-fragments).
  v16bf qf[4];
  {
    const __bf16* qp = q + ((size_t)(b * S_ + qrow0 + ln) * D_) + h * HD_;
    #pragma unroll
    for (int kc = 0; kc < 4; ++kc)
      qf[kc] = *(const v16bf*)(qp + kc * 32 + lg * 16);
  }

  v8f   ot[8];
  float rm[8], rs[8];
  #pragma unroll
  for (int t = 0; t < 8; ++t)
    #pragma unroll
    for (int e = 0; e < 8; ++e) ot[t][e] = 0.0f;
  #pragma unroll
  for (int r = 0; r < 8; ++r) { rm[r] = -1e30f; rs[r] = 0.0f; }

  const int nj = (qtile + 1) * 4;         // causal bound in 32-key blocks
  for (int j = 0; j < nj; ++j) {
    // Cooperative K/V stage. K: async raw copy. V: transpose into [hd][key].
    #pragma unroll
    for (int i = 0; i < 2; ++i) {
      int cid = tid + i * 256;            // 512 chunks of 8 halves
      int key = cid >> 4;
      int cc  = (cid & 15) << 3;
      size_t rowoff = (size_t)(b * S_ + j * 32 + key) * KVD_ + kvh * HD_ + cc;
#if USE_ASYNC
      __builtin_amdgcn_global_load_async_to_lds_b128(GAS(k + rowoff),
                                                     LAS(&Ks[key][cc]), 0, 0);
#else
      *(v8bf*)&Ks[key][cc] = *(const v8bf*)(k + rowoff);
#endif
      v8bf vv = *(const v8bf*)(v + rowoff);
      #pragma unroll
      for (int e = 0; e < 8; ++e) Vs[cc + e][key] = vv[e];
    }
#if USE_ASYNC
    __builtin_amdgcn_s_wait_asynccnt(0);
#endif
    __syncthreads();

    if (j * 32 <= qrow0 + 15) {           // wave-uniform causal early-out
      // Scores: 16 rows x 32 keys = two 16x16 f32 tiles, 4 WMMAs each.
      v8f c0, c1;
      #pragma unroll
      for (int e = 0; e < 8; ++e) { c0[e] = 0.0f; c1[e] = 0.0f; }
      #pragma unroll
      for (int kc = 0; kc < 4; ++kc) {
        v16bf b0 = *(const v16bf*)&Ks[ln][kc * 32 + lg * 16];
        v16bf b1 = *(const v16bf*)&Ks[16 + ln][kc * 32 + lg * 16];
        c0 = wmma_bf16(qf[kc], b0, c0);
        c1 = wmma_bf16(qf[kc], b1, c1);
      }

      const int key0 = j * 32 + ln;
      #pragma unroll
      for (int r = 0; r < 8; ++r) {
        int qg = qrow0 + r + 8 * lg;
        bool m0 = key0 > qg, m1 = key0 + 16 > qg;
        float s0 = m0 ? -1e30f : c0[r] * scale;
        float s1 = m1 ? -1e30f : c1[r] * scale;
        float mx = fmaxf(rm[r], rowmax16(fmaxf(s0, s1)));
        float corr = __expf(rm[r] - mx);
        float p0 = m0 ? 0.0f : __expf(s0 - mx);
        float p1 = m1 ? 0.0f : __expf(s1 - mx);
        rs[r] = rs[r] * corr + rowsum16(p0 + p1);
        rm[r] = mx;
        #pragma unroll
        for (int t = 0; t < 8; ++t) ot[t][r] *= corr;
        Ps[w][r + 8 * lg][ln]      = (__bf16)p0;
        Ps[w][r + 8 * lg][16 + ln] = (__bf16)p1;
      }

      // P (16x32 bf16) as A-fragment, V-tile as B: O += P @ V  (8 WMMAs).
      v16bf pf = *(const v16bf*)&Ps[w][ln][lg * 16];
      #pragma unroll
      for (int t = 0; t < 8; ++t) {
        v16bf bv = *(const v16bf*)&Vs[t * 16 + ln][lg * 16];
        ot[t] = wmma_bf16(pf, bv, ot[t]);
      }
    }
    __syncthreads();
  }

  // Normalize and store (bf16 attention output, [B,S,H*hd]).
  #pragma unroll
  for (int r = 0; r < 8; ++r) {
    float inv = 1.0f / rs[r];
    int row = qrow0 + r + 8 * lg;
    __bf16* op = o + ((size_t)(b * S_ + row) * D_) + h * HD_;
    #pragma unroll
    for (int t = 0; t < 8; ++t) op[t * 16 + ln] = (__bf16)(ot[t][r] * inv);
  }
}

// ---------------------------------------------------------------------------
// Launch: cvt x/w -> bf16, QKV GEMMs -> RoPE -> attention -> O GEMM.
// Workspace: qb(32M) kb(8M) vb(8M) ab(32M) xb(32M) wb(32M reused per weight).
// ---------------------------------------------------------------------------
extern "C" void kernel_launch(void* const* d_in, const int* in_sizes, int n_in,
                              void* d_out, int out_size, void* d_ws, size_t ws_size,
                              hipStream_t stream) {
  const float* x    = (const float*)d_in[0];
  const float* cosb = (const float*)d_in[1];
  const float* sinb = (const float*)d_in[2];
  const float* wq   = (const float*)d_in[3];
  const float* wk   = (const float*)d_in[4];
  const float* wv   = (const float*)d_in[5];
  const float* wo   = (const float*)d_in[6];

  const int M = B_ * S_;                       // 4096
  __bf16* qb = (__bf16*)d_ws;                  // [M, 4096]
  __bf16* kb = qb + (size_t)M * D_;            // [M, 1024]
  __bf16* vb = kb + (size_t)M * KVD_;          // [M, 1024]
  __bf16* ab = vb + (size_t)M * KVD_;          // [M, 4096]
  __bf16* xb = ab + (size_t)M * D_;            // [M, 4096]
  __bf16* wb = xb + (size_t)M * D_;            // [4096, 4096] scratch weight

  const int nXL = (M * D_) / 8;                // big (4096x4096) tensors
  const int nKV = (KVD_ * D_) / 8;             // 1024x4096 tensors
  const int CB = 256;

  cvt_bf16_kernel<<<(nXL + CB - 1) / CB, CB, 0, stream>>>(x, xb, nXL);

  cvt_bf16_kernel<<<(nXL + CB - 1) / CB, CB, 0, stream>>>(wq, wb, nXL);
  gemm_bf16_kernel<true><<<dim3(D_ / 128, M / 128), 256, 0, stream>>>(
      xb, wb, qb, M, D_, D_);

  cvt_bf16_kernel<<<(nKV + CB - 1) / CB, CB, 0, stream>>>(wk, wb, nKV);
  gemm_bf16_kernel<true><<<dim3(KVD_ / 128, M / 128), 256, 0, stream>>>(
      xb, wb, kb, M, KVD_, D_);

  cvt_bf16_kernel<<<(nKV + CB - 1) / CB, CB, 0, stream>>>(wv, wb, nKV);
  gemm_bf16_kernel<true><<<dim3(KVD_ / 128, M / 128), 256, 0, stream>>>(
      xb, wb, vb, M, KVD_, D_);

  rope_kernel<<<(M * (D_ / 2) + CB - 1) / CB, CB, 0, stream>>>(qb, cosb, sinb, M, D_);
  rope_kernel<<<(M * (KVD_ / 2) + CB - 1) / CB, CB, 0, stream>>>(kb, cosb, sinb, M, KVD_);

  attn_kernel<<<dim3(S_ / 128, H_, B_), 256, 0, stream>>>(qb, kb, vb, ab);

  cvt_bf16_kernel<<<(nXL + CB - 1) / CB, CB, 0, stream>>>(wo, wb, nXL);
  gemm_bf16_kernel<false><<<dim3(D_ / 128, M / 128), 256, 0, stream>>>(
      ab, wb, d_out, M, D_, D_);
}
